// PtrGenOutput_39496519254422
// MI455X (gfx1250) — matrix operand order repaired and checked by reading, not compiled, not run on hardware
//
#include <hip/hip_runtime.h>
#include <hip/hip_bf16.h>
#include <math.h>

// Problem constants (from reference)
constexpr int Bn    = 1024;   // batch
constexpr int Hd    = 1024;   // hidden
constexpr int SRC   = 64;
constexpr int V_IN  = 30000;
constexpr int V_OUT = 50000;

// Output layout (flat, f32): out_probs | ptr_or_gen | gen_probs | attn_probs
constexpr size_t O_OUT  = 0;
constexpr size_t O_PG   = (size_t)Bn * V_OUT;
constexpr size_t O_GEN  = O_PG + (size_t)Bn * 2;
constexpr size_t O_ATTN = O_GEN + (size_t)Bn * V_OUT;

typedef __attribute__((ext_vector_type(16))) __bf16 v16bf;
typedef __attribute__((ext_vector_type(8)))  __bf16 v8bf;
typedef __attribute__((ext_vector_type(4)))  __bf16 v4bf;
typedef __attribute__((ext_vector_type(8)))  float  v8f;
typedef __attribute__((ext_vector_type(4)))  float  v4f;
typedef __attribute__((ext_vector_type(4)))  int    v4i;

static __device__ __forceinline__ v16bf cat8(v8bf a, v8bf b) {
  return __builtin_shufflevector(a, b, 0,1,2,3,4,5,6,7,8,9,10,11,12,13,14,15);
}

// ---- CDNA5 async global->LDS path (guarded; sync bf16 fallback otherwise) --
#if __has_builtin(__builtin_amdgcn_global_load_async_to_lds_b128)
  #define USE_ASYNC 1
  #define GLOBAL_AS __attribute__((address_space(1)))
  #define LDS_AS    __attribute__((address_space(3)))
  #if __has_builtin(__builtin_amdgcn_s_wait_asynccnt)
    #define WAIT_ASYNC(n) __builtin_amdgcn_s_wait_asynccnt(n)
  #else
    #define WAIT_ASYNC(n) asm volatile("s_wait_asynccnt %0" :: "n"(n) : "memory")
  #endif
static __device__ __forceinline__ void async_cp16(const __bf16* g, __bf16* l) {
  // builtin signature (per hipcc diagnostic): (v4i AS1*, v4i AS3*, imm, imm)
  __builtin_amdgcn_global_load_async_to_lds_b128(
      (GLOBAL_AS v4i*)g, (LDS_AS v4i*)l, 0, 0);
}
#else
  #define USE_ASYNC 0
  #define WAIT_ASYNC(n) do {} while (0)
#endif

// ---------------------------------------------------------------------------
// K0: split x into bf16 hi/lo planes once (reused by all 196 N-block columns)
// ---------------------------------------------------------------------------
__global__ __launch_bounds__(256) void convert_x_kernel(const float* __restrict__ x,
                                                        __bf16* __restrict__ xhi,
                                                        __bf16* __restrict__ xlo) {
  int i = (blockIdx.x * 256 + threadIdx.x) * 4;
  v4f v = *(const v4f*)(x + i);
  v4bf h, l;
  for (int j = 0; j < 4; ++j) {
    float f = v[j];
    __bf16 hh = (__bf16)f;
    h[j] = hh;
    l[j] = (__bf16)(f - (float)hh);
  }
  *(v4bf*)(xhi + i) = h;
  *(v4bf*)(xlo + i) = l;
}

// ---------------------------------------------------------------------------
// K1: copy-or-gen gate: ptr_or_gen = log_softmax(x @ Wc.T + bc)  -> [B,2]
// ---------------------------------------------------------------------------
__global__ __launch_bounds__(256) void gate_kernel(const float* __restrict__ x,
                                                   const float* __restrict__ Wc,
                                                   const float* __restrict__ bc,
                                                   float* __restrict__ out_pg) {
  __shared__ float r0[256], r1[256];
  const int b = blockIdx.x, tid = threadIdx.x;
  const float* xr = x + (size_t)b * Hd;
  float a0 = 0.f, a1 = 0.f;
  for (int i = tid; i < Hd; i += 256) {
    float xv = xr[i];
    a0 += xv * Wc[i];
    a1 += xv * Wc[Hd + i];
  }
  r0[tid] = a0; r1[tid] = a1;
  __syncthreads();
  for (int st = 128; st > 0; st >>= 1) {
    if (tid < st) { r0[tid] += r0[tid + st]; r1[tid] += r1[tid + st]; }
    __syncthreads();
  }
  if (tid == 0) {
    float l0 = r0[0] + bc[0], l1 = r1[0] + bc[1];
    float mm = fmaxf(l0, l1);
    float ls = mm + logf(expf(l0 - mm) + expf(l1 - mm));
    out_pg[b * 2 + 0] = l0 - ls;
    out_pg[b * 2 + 1] = l1 - ls;
  }
}

// ---------------------------------------------------------------------------
// K2: attention softmax over SRC=64 -> [B,64]
// ---------------------------------------------------------------------------
__global__ __launch_bounds__(64) void attn_kernel(const float* __restrict__ attn,
                                                  float* __restrict__ out_ap) {
  __shared__ float buf[64];
  const int b = blockIdx.x, tid = threadIdx.x;
  float e = attn[b * SRC + tid];
  buf[tid] = e;
  __syncthreads();
  for (int st = 32; st > 0; st >>= 1) {
    if (tid < st) buf[tid] = fmaxf(buf[tid], buf[tid + st]);
    __syncthreads();
  }
  float mx = buf[0];
  __syncthreads();
  float ex = expf(e - mx);
  buf[tid] = ex;
  __syncthreads();
  for (int st = 32; st > 0; st >>= 1) {
    if (tid < st) buf[tid] += buf[tid + st];
    __syncthreads();
  }
  out_ap[b * SRC + tid] = ex / buf[0];
}

// ---------------------------------------------------------------------------
// K3: vout_src[v] = max j with inp_to_act[j]==v, else -1 (last-writer-wins
// model of jax .at[:, idx].set with duplicate indices)
// ---------------------------------------------------------------------------
__global__ void vout_init(int* __restrict__ vout) {
  int v = blockIdx.x * 256 + threadIdx.x;
  if (v < V_OUT) vout[v] = -1;
}
__global__ void vout_scat(const int* __restrict__ inp_to_act, int* __restrict__ vout) {
  int j = blockIdx.x * 256 + threadIdx.x;
  if (j < V_IN) atomicMax(&vout[inp_to_act[j]], j);
}

// ---------------------------------------------------------------------------
// K4: GEMM  gen_logits[b, v] = sum_h x[b,h] * Wg[v,h] + bg[v]
// bf16x3 split-precision via V_WMMA_F32_16X16X32_BF16.
// 256 thr (8 waves, 2Mx4N), macro-tile 64M x 256N, K-step 32,
// double-buffered LDS; A staged via GLOBAL_LOAD_ASYNC_TO_LDS_B128 (ASYNCcnt),
// B f32 software-pipelined through VGPRs with on-the-fly hi/lo split.
// ---------------------------------------------------------------------------
constexpr int BM = 64;
constexpr int BN = 256;
constexpr int BK = 32;
constexpr int KSTEPS = Hd / BK;   // 32

__global__ __launch_bounds__(256) void gemm_bf16x3_kernel(
    const __bf16* __restrict__ xhi, const __bf16* __restrict__ xlo,
    const float* __restrict__ Wg, const float* __restrict__ bg,
    float* __restrict__ out_logits) {
  __shared__ __bf16 As_hi[2][BM][BK];
  __shared__ __bf16 As_lo[2][BM][BK];
  __shared__ __bf16 Bs_hi[2][BN][BK];
  __shared__ __bf16 Bs_lo[2][BN][BK];

  const int tid  = threadIdx.x;
  const int lane = tid & 31;
  const int w    = tid >> 5;
  const int wm   = w >> 2;       // 0..1
  const int wn   = w & 3;        // 0..3
  const int m0   = blockIdx.y * BM;
  const int n0   = blockIdx.x * BN;

  v8f acc[2][4];
  for (int mi = 0; mi < 2; ++mi)
    for (int ni = 0; ni < 4; ++ni)
      acc[mi][ni] = (v8f){0.f,0.f,0.f,0.f,0.f,0.f,0.f,0.f};

  const int rsel = lane & 15;
  const int kh   = lane >> 4;

  // ---- A staging: 256 threads cover 64 rows x 4 chunks of 8 bf16 (16B) ----
  const int arow = tid >> 2;
  const int ac8  = (tid & 3) * 8;
  auto issueA = [&](int k0, int buf) {
    const __bf16* gh = xhi + (size_t)(m0 + arow) * Hd + k0 + ac8;
    const __bf16* gl = xlo + (size_t)(m0 + arow) * Hd + k0 + ac8;
#if USE_ASYNC
    async_cp16(gh, &As_hi[buf][arow][ac8]);
    async_cp16(gl, &As_lo[buf][arow][ac8]);
#else
    *(v8bf*)&As_hi[buf][arow][ac8] = *(const v8bf*)gh;
    *(v8bf*)&As_lo[buf][arow][ac8] = *(const v8bf*)gl;
#endif
  };

  // ---- B staging: 2048 float4 slots (256 rows x 8 chunks) -----------------
  auto loadB = [&](v4f* r, int k0) {
    for (int i = 0; i < 8; ++i) {
      int slot = tid + i * 256;
      int row = slot >> 3, c4 = (slot & 7) << 2;
      int n = n0 + row;
      r[i] = (v4f){0.f, 0.f, 0.f, 0.f};
      if (n < V_OUT) {
        const float* gp = Wg + (size_t)n * Hd + k0 + c4;
        r[i] = *(const v4f*)gp;
        if (k0 + BK < Hd) __builtin_prefetch(gp + BK, 0, 0);
      }
    }
  };
  auto storeB = [&](const v4f* r, int buf) {
    for (int i = 0; i < 8; ++i) {
      int slot = tid + i * 256;
      int row = slot >> 3, c4 = (slot & 7) << 2;
      v4bf h, l;
      for (int j = 0; j < 4; ++j) {
        float f = r[i][j];
        __bf16 hh = (__bf16)f;
        h[j] = hh;
        l[j] = (__bf16)(f - (float)hh);
      }
      *(v4bf*)&Bs_hi[buf][row][c4] = h;
      *(v4bf*)&Bs_lo[buf][row][c4] = l;
    }
  };

  auto compute = [&](int buf) {
    // B fragments (32x16 bf16): lanes0-15 K=0..15, lanes16-31 K=16..31
    v16bf bh[4], bl[4];
    for (int ni = 0; ni < 4; ++ni) {
      int n = wn * 64 + ni * 16 + rsel;
      const v8bf* ph = (const v8bf*)&Bs_hi[buf][n][kh * 16];
      const v8bf* pl = (const v8bf*)&Bs_lo[buf][n][kh * 16];
      bh[ni] = cat8(ph[0], ph[1]);
      bl[ni] = cat8(pl[0], pl[1]);
    }
    // A fragments (16x32 bf16): lane half kh -> K range kh*16..kh*16+15
    for (int mi = 0; mi < 2; ++mi) {
      int m = wm * 32 + mi * 16 + rsel;
      v16bf ah = cat8(*(const v8bf*)&As_hi[buf][m][kh * 16],
                      *(const v8bf*)&As_hi[buf][m][kh * 16 + 8]);
      v16bf al = cat8(*(const v8bf*)&As_lo[buf][m][kh * 16],
                      *(const v8bf*)&As_lo[buf][m][kh * 16 + 8]);
      for (int ni = 0; ni < 4; ++ni) {
        acc[mi][ni] = __builtin_amdgcn_wmma_f32_16x16x32_bf16(
            false, ah, false, bh[ni], (short)0, acc[mi][ni], false, false);
        acc[mi][ni] = __builtin_amdgcn_wmma_f32_16x16x32_bf16(
            false, ah, false, bl[ni], (short)0, acc[mi][ni], false, false);
        acc[mi][ni] = __builtin_amdgcn_wmma_f32_16x16x32_bf16(
            false, al, false, bh[ni], (short)0, acc[mi][ni], false, false);
      }
    }
  };

  // ---- software-pipelined main loop ---------------------------------------
  v4f regB[8];
  loadB(regB, 0);
  issueA(0, 0);
  for (int s = 0; s < KSTEPS; ++s) {
    const int buf = s & 1;
    storeB(regB, buf);                 // consumes stage-s B regs
    if (s + 1 < KSTEPS) {
      loadB(regB, (s + 1) * BK);       // stage s+1 B in flight
      issueA((s + 1) * BK, buf ^ 1);   // stage s+1 A async in flight
      WAIT_ASYNC(2);                   // retire stage-s A copies (in-order)
    } else {
      WAIT_ASYNC(0);
    }
    __syncthreads();
    compute(buf);
    __syncthreads();
  }

  // ---- epilogue: C/D 16x16: VGPR r -> M = r + 8*(lane>=16), N = lane&15 ---
  for (int ni = 0; ni < 4; ++ni) {
    int n = n0 + wn * 64 + ni * 16 + rsel;
    if (n >= V_OUT) continue;
    float bias = bg[n];
    for (int mi = 0; mi < 2; ++mi) {
      int mbase = m0 + wm * 32 + mi * 16 + kh * 8;
      for (int r = 0; r < 8; ++r)
        out_logits[(size_t)(mbase + r) * V_OUT + n] = acc[mi][ni][r] + bias;
    }
  }
}

// ---------------------------------------------------------------------------
// K5: per-row online log-sum-exp over V_OUT
// ---------------------------------------------------------------------------
__global__ __launch_bounds__(256) void lse_kernel(const float* __restrict__ gen,
                                                  float* __restrict__ row_m,
                                                  float* __restrict__ row_ls) {
  __shared__ float sm[256], ss[256];
  const int b = blockIdx.x, tid = threadIdx.x;
  const float* rowp = gen + (size_t)b * V_OUT;
  float m = -INFINITY, s = 0.f;
  for (int v = tid; v < V_OUT; v += 256) {
    float l  = rowp[v];
    float nm = fmaxf(m, l);
    s = s * expf(m - nm) + expf(l - nm);
    m = nm;
  }
  sm[tid] = m; ss[tid] = s;
  __syncthreads();
  for (int st = 128; st > 0; st >>= 1) {
    if (tid < st) {
      float m2 = sm[tid + st], s2 = ss[tid + st];
      float nm = fmaxf(sm[tid], m2);
      ss[tid] = ss[tid] * expf(sm[tid] - nm) + s2 * expf(m2 - nm);
      sm[tid] = nm;
    }
    __syncthreads();
  }
  if (tid == 0) { row_m[b] = sm[0]; row_ls[b] = logf(ss[0]); }
}

// ---------------------------------------------------------------------------
// K6: fused epilogue. Per row b:
//   - build inpdist[V_IN] in LDS (120 KB; CDNA5 WGP has 320 KB LDS)
//   - gen_probs = logits - m - lse (in place)
//   - ptr_probs = log(inpdist[vout_src[v]]) (or log(0) = -inf)
//   - out = pg0*gen_probs + pg1*ptr_probs; -0 squashed to +0
// ---------------------------------------------------------------------------
__global__ __launch_bounds__(256) void final_kernel(const int* __restrict__ ctx_ids,
                                                    const int* __restrict__ vout_src,
                                                    const float* __restrict__ row_m,
                                                    const float* __restrict__ row_ls,
                                                    float* __restrict__ out) {
  __shared__ float inpdist[V_IN];  // 120000 bytes of LDS
  const int b = blockIdx.x, tid = threadIdx.x;
  for (int i = tid; i < V_IN; i += 256) inpdist[i] = 0.f;
  __syncthreads();
  if (tid < SRC) {
    int j = ctx_ids[b * SRC + tid];
    atomicAdd(&inpdist[j], out[O_ATTN + b * SRC + tid]);
  }
  __syncthreads();

  const float pg0 = out[O_PG + b * 2 + 0];
  const float pg1 = out[O_PG + b * 2 + 1];
  const float m = row_m[b], ls = row_ls[b];
  float* genrow = out + O_GEN + (size_t)b * V_OUT;
  float* outrow = out + O_OUT + (size_t)b * V_OUT;

  for (int v = tid; v < V_OUT; v += 256) {
    float gl = genrow[v];
    float gp = gl - m - ls;
    int   j  = vout_src[v];
    float sc = (j >= 0) ? inpdist[j] : 0.f;
    float pp = logf(sc);               // log(0) = -inf, as in reference
    float o  = pg0 * gp + pg1 * pp;
    o = (o == 0.f) ? 0.f : o;          // squash -0 like jnp.where(out==0, 0, out)
    genrow[v] = gp;
    outrow[v] = o;
  }
}

// ---------------------------------------------------------------------------
static constexpr size_t align256(size_t x) { return (x + 255) & ~(size_t)255; }

extern "C" void kernel_launch(void* const* d_in, const int* in_sizes, int n_in,
                              void* d_out, int out_size, void* d_ws, size_t ws_size,
                              hipStream_t stream) {
  const float* x          = (const float*)d_in[0];
  const float* attn       = (const float*)d_in[1];
  const float* Wg         = (const float*)d_in[2];
  const float* bg         = (const float*)d_in[3];
  const float* Wc         = (const float*)d_in[4];
  const float* bc         = (const float*)d_in[5];
  const int*   ctx_ids    = (const int*)d_in[6];
  const int*   inp_to_act = (const int*)d_in[7];
  // d_in[8] = out_map is the identity; gather skipped.

  float* out = (float*)d_out;

  // ws layout: vout_src | row_m | row_ls | xhi | xlo   (~4.4 MB total)
  char* wsb = (char*)d_ws;
  size_t off = 0;
  int*    vout_src = (int*)(wsb + off);   off += align256((size_t)V_OUT * sizeof(int));
  float*  row_m    = (float*)(wsb + off); off += align256((size_t)Bn * sizeof(float));
  float*  row_ls   = (float*)(wsb + off); off += align256((size_t)Bn * sizeof(float));
  __bf16* xhi      = (__bf16*)(wsb + off); off += align256((size_t)Bn * Hd * sizeof(__bf16));
  __bf16* xlo      = (__bf16*)(wsb + off);

  convert_x_kernel<<<(Bn * Hd) / (256 * 4), 256, 0, stream>>>(x, xhi, xlo);
  gate_kernel<<<Bn, 256, 0, stream>>>(x, Wc, bc, out + O_PG);
  attn_kernel<<<Bn, 64, 0, stream>>>(attn, out + O_ATTN);
  vout_init<<<(V_OUT + 255) / 256, 256, 0, stream>>>(vout_src);
  vout_scat<<<(V_IN + 255) / 256, 256, 0, stream>>>(inp_to_act, vout_src);

  dim3 grid((V_OUT + BN - 1) / BN, Bn / BM);   // 196 x 16
  gemm_bf16x3_kernel<<<grid, 256, 0, stream>>>(xhi, xlo, Wg, bg, out + O_GEN);

  lse_kernel<<<Bn, 256, 0, stream>>>(out + O_GEN, row_m, row_ls);
  final_kernel<<<Bn, 256, 0, stream>>>(ctx_ids, vout_src, row_m, row_ls, out);
}
